// VehicleReconstructor_63385127355047
// MI455X (gfx1250) — matrix operand order repaired and checked by reading, not compiled, not run on hardware
//
#include <hip/hip_runtime.h>
#include <hip/hip_bf16.h>

// ---------------------------------------------------------------------------
// MLA attention + wheel projections for MI455X (gfx1250, wave32, WMMA).
// All GEMMs run on v_wmma_f32_16x16x32_bf16 (f32 accumulate).
// Global->LDS tile movement uses the Tensor Data Mover (tensor_load_to_lds +
// s_wait_tensorcnt) in both the GEMM and the flash-attention kernels, double
// buffered so the DMA overlaps WMMA compute (1 barrier per tile step).
// Weights pre-transposed to [N][K] bf16; V pre-transposed to [1024][S].
// ---------------------------------------------------------------------------

typedef __bf16 bf16;
typedef __attribute__((ext_vector_type(16))) __bf16 v16bf;
typedef __attribute__((ext_vector_type(8)))  __bf16 v8bf;
typedef __attribute__((ext_vector_type(4)))  __bf16 v4bf;
typedef __attribute__((ext_vector_type(8)))  float  v8f;

typedef unsigned int u32;
typedef __attribute__((ext_vector_type(4))) unsigned int u32x4;
typedef __attribute__((ext_vector_type(8))) int          i32x8;
typedef __attribute__((ext_vector_type(4))) int          i32x4;

#define S_LEN 2048
#define DM    1024
#define NH    16
#define HD    64
#define LAT   256

#if defined(__has_builtin)
#if __has_builtin(__builtin_amdgcn_tensor_load_to_lds) && \
    __has_builtin(__builtin_amdgcn_s_wait_tensorcnt)
#define USE_TDM 1
#endif
#endif
#ifndef USE_TDM
#define USE_TDM 0
#endif

static __device__ __forceinline__ v16bf cat8(v8bf lo, v8bf hi) {
  return __builtin_shufflevector(lo, hi, 0,1,2,3,4,5,6,7,8,9,10,11,12,13,14,15);
}

static __device__ __forceinline__ v8f wmma_bf16(v16bf a, v16bf b, v8f c) {
  return __builtin_amdgcn_wmma_f32_16x16x32_bf16(false, a, false, b, (short)0, c,
                                                 false, false);
}

#if USE_TDM
// Issue one TDM 2D tile load: tile_h x tile_w bf16 elements from a row-major
// tensor (row stride = stride_elems) into contiguous LDS at lds_off.
// D# per CDNA5 ISA ch.8: group0 {count, lds_addr, global_addr, type=2},
// group1 {data_size=1(2B), tensor dims, tile dims, dim0 stride}, groups 2/3=0.
static __device__ __forceinline__ void
tdm_load_2d(const bf16* gsrc, u32 lds_off,
            u32 tensor_w, u32 tensor_h, u32 stride_elems,
            u32 tile_w, u32 tile_h)
{
  unsigned long long ga = (unsigned long long)(uintptr_t)gsrc;
  u32x4 g0;
  g0[0] = 1u;                                   // count=1 (valid), user mode
  g0[1] = lds_off;                              // LDS byte address
  g0[2] = (u32)ga;                              // global_addr[31:0]
  g0[3] = ((u32)(ga >> 32) & 0x01FFFFFFu) | (2u << 30);  // addr[56:32] | type=2
  i32x8 g1;
  g1[0] = (int)(1u << 16);                      // data_size=1 -> 2 bytes/elem
  g1[1] = (int)((tensor_w & 0xFFFFu) << 16);    // tensor_dim0[15:0]
  g1[2] = (int)((tensor_w >> 16) | ((tensor_h & 0xFFFFu) << 16));
  g1[3] = (int)((tensor_h >> 16) | ((tile_w & 0xFFFFu) << 16));  // tile_dim0
  g1[4] = (int)(tile_h & 0xFFFFu);              // tile_dim1 (tile_dim2=0)
  g1[5] = (int)stride_elems;                    // tensor_dim0_stride[31:0]
  g1[6] = 0;                                    // stride hi / dim1_stride lo
  g1[7] = 0;
  i32x4 z4 = {0, 0, 0, 0};
#if __clang_major__ >= 23
  i32x8 z8 = {0, 0, 0, 0, 0, 0, 0, 0};
  __builtin_amdgcn_tensor_load_to_lds(g0, g1, z4, z4, z8, 0);
#else
  __builtin_amdgcn_tensor_load_to_lds(g0, g1, z4, z4, 0);
#endif
}
#endif  // USE_TDM

// ------------------------------- f32 -> bf16 -------------------------------
__global__ void cvt_f32_bf16(const float* __restrict__ in, bf16* __restrict__ out, int n) {
  int i = blockIdx.x * blockDim.x + threadIdx.x;
  int stride = gridDim.x * blockDim.x;
  for (; i < n; i += stride) out[i] = (bf16)in[i];
}

// -------------------- f32 [R][C] -> bf16 [C][R] transpose ------------------
__global__ __launch_bounds__(256)
void transpose_f32_bf16(const float* __restrict__ in, bf16* __restrict__ out,
                        int R, int C) {
  __shared__ float t[32][33];
  const int r0 = blockIdx.y * 32, c0 = blockIdx.x * 32;
  const int tx = threadIdx.x, ty = threadIdx.y;
  #pragma unroll
  for (int j = 0; j < 4; ++j) {
    int r = ty + j * 8;
    t[r][tx] = in[(size_t)(r0 + r) * C + c0 + tx];
  }
  __syncthreads();
  #pragma unroll
  for (int j = 0; j < 4; ++j) {
    int r = ty + j * 8;
    out[(size_t)(c0 + r) * R + r0 + tx] = (bf16)t[tx][r];
  }
}

// -------------------- bf16 [R][C] -> bf16 [C][R] transpose -----------------
__global__ __launch_bounds__(256)
void transpose_bf16(const bf16* __restrict__ in, bf16* __restrict__ out,
                    int R, int C) {
  __shared__ bf16 t[32][34];
  const int r0 = blockIdx.y * 32, c0 = blockIdx.x * 32;
  const int tx = threadIdx.x, ty = threadIdx.y;
  #pragma unroll
  for (int j = 0; j < 4; ++j) {
    int r = ty + j * 8;
    t[r][tx] = in[(size_t)(r0 + r) * C + c0 + tx];
  }
  __syncthreads();
  #pragma unroll
  for (int j = 0; j < 4; ++j) {
    int r = ty + j * 8;
    out[(size_t)(c0 + r) * R + r0 + tx] = t[tx][r];
  }
}

// -------------------------- generic bf16 WMMA GEMM -------------------------
// C[M,N] = A[M,K] @ Bt[N,K]^T (+bias[N]).  256 threads = 8 waves laid out as
// 4 (M) x 2 (N); each wave computes a 32x64 strip = 2x4 WMMA accumulators.
// LDS double buffered: A tile 128x32 [m][k], B tile 128x32 [n][k].
// Tiles DMA'd by TDM (wave 0 issues, TENSORcnt) or register-staged fallback.
// Dims: M,N multiples of 128; K multiple of 32.
__global__ __launch_bounds__(256)
void gemm_bf16_128x128(const bf16* __restrict__ A, const bf16* __restrict__ Bt,
                       int M, int N, int K,
                       float* __restrict__ Cf, int ldc,
                       bf16*  __restrict__ Cb, int ldcb,
                       const float* __restrict__ bias)
{
  __shared__ bf16 a_t[2][128][32];   // [buf][m][k]
  __shared__ bf16 b_t[2][128][32];   // [buf][n][k]  (K-contiguous per column)

  const int tid    = threadIdx.x;
  const int wave   = tid >> 5;
  const int lane   = tid & 31;
  const int lanelo = lane & 15;
  const int lhalf  = lane >> 4;
  const int waveM  = wave & 3;        // 0..3 -> 32-row strip
  const int waveN  = wave >> 2;       // 0..1 -> 64-col half
  const int bm = blockIdx.y * 128;
  const int bn = blockIdx.x * 128;

  v8f acc[2][4];
  const v8f vzero = {0.f,0.f,0.f,0.f,0.f,0.f,0.f,0.f};
  #pragma unroll
  for (int m = 0; m < 2; ++m)
    #pragma unroll
    for (int t = 0; t < 4; ++t) acc[m][t] = vzero;

#if USE_TDM
  auto issue = [&](int buf, int kk) {
    tdm_load_2d(A  + (size_t)bm * K + kk, (u32)(uintptr_t)&a_t[buf][0][0],
                (u32)K, (u32)M, (u32)K, 32u, 128u);
    tdm_load_2d(Bt + (size_t)bn * K + kk, (u32)(uintptr_t)&b_t[buf][0][0],
                (u32)K, (u32)N, (u32)K, 32u, 128u);
  };
  if (wave == 0) {
    issue(0, 0);
    __builtin_amdgcn_s_wait_tensorcnt(0);
  }
  __syncthreads();
#else
  v4bf arg[4], brg[4];                // staged global loads
  auto stage = [&](int kk) {
    #pragma unroll
    for (int j = 0; j < 4; ++j) {
      int i   = tid + 256 * j;        // 0..1023, 4 elems each
      int row = i >> 3;               // 0..127
      int c4  = (i & 7) * 4;          // 0..28
      arg[j] = *(const v4bf*)(A  + (size_t)(bm + row) * K + kk + c4);
      brg[j] = *(const v4bf*)(Bt + (size_t)(bn + row) * K + kk + c4);
      if (j == 0 && kk + 64 < K) {    // prefetch 2 k-steps ahead (WGP$)
        __builtin_prefetch(A  + (size_t)(bm + row) * K + kk + 64 + c4, 0, 3);
        __builtin_prefetch(Bt + (size_t)(bn + row) * K + kk + 64 + c4, 0, 3);
      }
    }
  };
  auto commit = [&](int buf) {
    #pragma unroll
    for (int j = 0; j < 4; ++j) {
      int i   = tid + 256 * j;
      int row = i >> 3;
      int c4  = (i & 7) * 4;
      *(v4bf*)&a_t[buf][row][c4] = arg[j];
      *(v4bf*)&b_t[buf][row][c4] = brg[j];
    }
  };
  stage(0);
  commit(0);
  __syncthreads();
#endif

  const int nk = K >> 5;
  for (int step = 0; step < nk; ++step) {
    const int cur  = step & 1;
    const bool more = (step + 1) < nk;
#if USE_TDM
    if (more && wave == 0) issue(cur ^ 1, (step + 1) << 5);
#else
    if (more) stage((step + 1) << 5);
#endif

    // A fragments: row = lane%16; 16 contiguous K at (lane<16?0:16)
    v16bf af[2];
    #pragma unroll
    for (int m = 0; m < 2; ++m) {
      int row  = waveM * 32 + m * 16 + lanelo;
      int aoff = lhalf * 16;
      af[m] = *(const v16bf*)&a_t[cur][row][aoff];
    }
    // B fragments: column = lane%16, 16 contiguous K at (lane<16?0:16)
    v16bf bf4[4];
    #pragma unroll
    for (int t = 0; t < 4; ++t)
      bf4[t] = *(const v16bf*)&b_t[cur][waveN * 64 + t * 16 + lanelo][lhalf * 16];

    #pragma unroll
    for (int m = 0; m < 2; ++m)
      #pragma unroll
      for (int t = 0; t < 4; ++t)
        acc[m][t] = wmma_bf16(af[m], bf4[t], acc[m][t]);

#if USE_TDM
    if (more && wave == 0) __builtin_amdgcn_s_wait_tensorcnt(0);
#else
    if (more) commit(cur ^ 1);
#endif
    __syncthreads();
  }

  // epilogue: C layout row = r + 8*lhalf within each 16x16 tile
  #pragma unroll
  for (int m = 0; m < 2; ++m) {
    #pragma unroll
    for (int t = 0; t < 4; ++t) {
      int col = bn + waveN * 64 + t * 16 + lanelo;
      float bv = bias ? bias[col] : 0.f;
      #pragma unroll
      for (int r = 0; r < 8; ++r) {
        int row = bm + waveM * 32 + m * 16 + r + 8 * lhalf;
        float v = acc[m][t][r] + bv;
        if (Cf) Cf[(size_t)row * ldc + col] = v;
        if (Cb) Cb[(size_t)row * ldcb + col] = (bf16)v;
      }
    }
  }
}

// --------------------------- flash MLA attention ---------------------------
// One block = (head h, 64-query tile). 4 waves; wave w owns 16 query rows.
// Iterates 32 key tiles of 64: S=Q*K^T via WMMA, online softmax in the WMMA
// C-layout, P->LDS(per-wave), O+=P*V via WMMA. K/V tiles DMA'd by the TDM,
// double buffered (issue buf^1, compute buf, wait TENSORcnt, 1 barrier/iter).
// V comes pre-transposed ([1024][S]). Output: bf16 ctx.
__global__ __launch_bounds__(128)
void mla_flash_attn(const bf16* __restrict__ Qg, const bf16* __restrict__ Kg,
                    const bf16* __restrict__ Vt, bf16* __restrict__ ctx)
{
  __shared__ bf16 lds_q[64][64];        // [q][d]
  __shared__ bf16 lds_k[2][64][64];     // [buf][key][d]
  __shared__ bf16 lds_vt[2][64][64];    // [buf][d][key]
  __shared__ bf16 lds_p[4][16][64];     // per-wave P tile [q][key]

  const int tid    = threadIdx.x;
  const int wave   = tid >> 5;
  const int lane   = tid & 31;
  const int lanelo = lane & 15;
  const int lhalf  = lane >> 4;
  const int qb   = blockIdx.x;          // 0..31
  const int h    = blockIdx.y;          // 0..15
  const int col0 = h * HD;

  // load Q tile once (vectorized; visible after first barrier)
  #pragma unroll
  for (int j = 0; j < 8; ++j) {
    int i = tid + 128 * j;              // 0..1023, 4 elems each
    int row = i >> 4;
    int c   = (i & 15) * 4;
    *(v4bf*)&lds_q[row][c] =
        *(const v4bf*)(Qg + (size_t)(qb * 64 + row) * DM + col0 + c);
  }

  const v8f vzero = {0.f,0.f,0.f,0.f,0.f,0.f,0.f,0.f};
  v8f acc_o[4];
  #pragma unroll
  for (int t = 0; t < 4; ++t) acc_o[t] = vzero;
  float mst[8], lst[8];
  #pragma unroll
  for (int r = 0; r < 8; ++r) { mst[r] = -1e30f; lst[r] = 0.f; }

  const float scale = 0.125f;           // 1/sqrt(64)
  const int nkb = S_LEN / 64;

#if USE_TDM
  auto issueKV = [&](int buf, int kb) {
    tdm_load_2d(Kg + (size_t)(kb * 64) * DM + col0,
                (u32)(uintptr_t)&lds_k[buf][0][0],
                (u32)DM, (u32)S_LEN, (u32)DM, 64u, 64u);
    tdm_load_2d(Vt + (size_t)col0 * S_LEN + kb * 64,
                (u32)(uintptr_t)&lds_vt[buf][0][0],
                (u32)S_LEN, (u32)DM, (u32)S_LEN, 64u, 64u);
  };
  if (wave == 0) {
    issueKV(0, 0);
    __builtin_amdgcn_s_wait_tensorcnt(0);
  }
  __syncthreads();
#endif

  for (int kb = 0; kb < nkb; ++kb) {
#if USE_TDM
    const int cur = kb & 1;
    const bool more = (kb + 1) < nkb;
    if (more && wave == 0) issueKV(cur ^ 1, kb + 1);
#else
    const int cur = 0;
    __syncthreads();                    // previous iteration's LDS reads done
    #pragma unroll
    for (int j = 0; j < 8; ++j) {
      int i = tid + 128 * j;
      int row = i >> 4;                 // K: key row / Vt: head-dim row
      int c   = (i & 15) * 4;
      *(v4bf*)&lds_k[0][row][c] =
          *(const v4bf*)(Kg + (size_t)(kb * 64 + row) * DM + col0 + c);
      *(v4bf*)&lds_vt[0][row][c] =
          *(const v4bf*)(Vt + (size_t)(col0 + row) * S_LEN + kb * 64 + c);
    }
    __syncthreads();
#endif

    // --- S = Q * K^T (16 queries x 64 keys per wave) ---
    v8f sA[4];
    #pragma unroll
    for (int t = 0; t < 4; ++t) sA[t] = vzero;
    const int arow = wave * 16 + lanelo;
    #pragma unroll
    for (int kk = 0; kk < 64; kk += 32) {
      int aoff = kk + lhalf * 16;
      v16bf afrag = *(const v16bf*)&lds_q[arow][aoff];
      #pragma unroll
      for (int t = 0; t < 4; ++t) {
        v16bf bfrag = *(const v16bf*)&lds_k[cur][t * 16 + lanelo][kk + lhalf * 16];
        sA[t] = wmma_bf16(afrag, bfrag, sA[t]);
      }
    }

    // --- online softmax on C layout; rows reduce over 16-lane halves ---
    #pragma unroll
    for (int r = 0; r < 8; ++r) {
      float s0 = sA[0][r] * scale, s1 = sA[1][r] * scale,
            s2 = sA[2][r] * scale, s3 = sA[3][r] * scale;
      float rowm = fmaxf(fmaxf(s0, s1), fmaxf(s2, s3));
      rowm = fmaxf(rowm, __shfl_xor(rowm, 1, 32));
      rowm = fmaxf(rowm, __shfl_xor(rowm, 2, 32));
      rowm = fmaxf(rowm, __shfl_xor(rowm, 4, 32));
      rowm = fmaxf(rowm, __shfl_xor(rowm, 8, 32));
      float mnew = fmaxf(mst[r], rowm);
      float corr = __expf(mst[r] - mnew);
      mst[r] = mnew;
      float p0 = __expf(s0 - mnew), p1 = __expf(s1 - mnew);
      float p2 = __expf(s2 - mnew), p3 = __expf(s3 - mnew);
      float rsum = p0 + p1 + p2 + p3;
      rsum += __shfl_xor(rsum, 1, 32);
      rsum += __shfl_xor(rsum, 2, 32);
      rsum += __shfl_xor(rsum, 4, 32);
      rsum += __shfl_xor(rsum, 8, 32);
      lst[r] = lst[r] * corr + rsum;
      acc_o[0][r] *= corr; acc_o[1][r] *= corr;
      acc_o[2][r] *= corr; acc_o[3][r] *= corr;
      int prow = r + 8 * lhalf;
      lds_p[wave][prow][ 0 + lanelo] = (bf16)p0;
      lds_p[wave][prow][16 + lanelo] = (bf16)p1;
      lds_p[wave][prow][32 + lanelo] = (bf16)p2;
      lds_p[wave][prow][48 + lanelo] = (bf16)p3;
    }
    // per-wave P region; same-wave LDS ops are in order -> no barrier needed

    // --- O += P * V ---
    #pragma unroll
    for (int kk = 0; kk < 64; kk += 32) {
      int aoff = kk + lhalf * 16;
      v16bf pfrag = *(const v16bf*)&lds_p[wave][lanelo][aoff];
      #pragma unroll
      for (int t = 0; t < 4; ++t) {
        v16bf vfrag = *(const v16bf*)&lds_vt[cur][t * 16 + lanelo][kk + lhalf * 16];
        acc_o[t] = wmma_bf16(pfrag, vfrag, acc_o[t]);
      }
    }

#if USE_TDM
    if (more && wave == 0) __builtin_amdgcn_s_wait_tensorcnt(0);
    __syncthreads();
#endif
  }

  // finalize: divide by row sums, store bf16 ctx
  #pragma unroll
  for (int r = 0; r < 8; ++r) {
    float inv = 1.f / lst[r];
    int row = qb * 64 + wave * 16 + r + 8 * lhalf;
    #pragma unroll
    for (int t = 0; t < 4; ++t) {
      ctx[(size_t)row * DM + col0 + t * 16 + lanelo] = (bf16)(acc_o[t][r] * inv);
    }
  }
}

// ------------------------------- launcher ----------------------------------
extern "C" void kernel_launch(void* const* d_in, const int* in_sizes, int n_in,
                              void* d_out, int out_size, void* d_ws, size_t ws_size,
                              hipStream_t stream) {
  const float* emb     = (const float*)d_in[0];   // [2048,1024]
  const float* Wq      = (const float*)d_in[1];   // [1024,1024]
  const float* Wdkv    = (const float*)d_in[2];   // [1024,256]
  const float* Wuk     = (const float*)d_in[3];   // [256,1024]
  const float* Wuv     = (const float*)d_in[4];   // [256,1024]
  const float* Wo      = (const float*)d_in[5];   // [1024,1024]
  const float* wheelW  = (const float*)d_in[6];   // [4,1024,256]
  const float* wheelB  = (const float*)d_in[7];   // [4,256]
  float* out = (float*)d_out;                     // [2048,2048]

  char* ws = (char*)d_ws;
  size_t off = 0;
  auto take = [&](size_t elems) -> bf16* {
    bf16* p = (bf16*)(ws + off);
    off += elems * sizeof(bf16);
    return p;
  };
  // weights stored transposed: [N][K]
  bf16* emb_bf  = take((size_t)S_LEN * DM);
  bf16* WqT     = take((size_t)DM * DM);      // [1024][1024]
  bf16* WdkvT   = take((size_t)LAT * DM);     // [256][1024]
  bf16* WukT    = take((size_t)DM * LAT);     // [1024][256]
  bf16* WuvT    = take((size_t)DM * LAT);     // [1024][256]
  bf16* WoT     = take((size_t)DM * DM);      // [1024][1024]
  bf16* whWT    = take((size_t)4 * 256 * DM); // [4][256][1024]
  bf16* q_bf    = take((size_t)S_LEN * DM);
  bf16* ckv_bf  = take((size_t)S_LEN * LAT);
  bf16* k_bf    = take((size_t)S_LEN * DM);
  bf16* v_bf    = take((size_t)S_LEN * DM);
  bf16* vt_bf   = take((size_t)DM * S_LEN);   // [1024][2048]
  bf16* ctx_bf  = take((size_t)S_LEN * DM);
  bf16* body_bf = take((size_t)S_LEN * DM);
  (void)ws_size; (void)n_in; (void)in_sizes; (void)out_size;

  // emb: plain convert (A operand, row-major)
  cvt_f32_bf16<<<(S_LEN * DM + 1023) / 1024, 256, 0, stream>>>(emb, emb_bf, S_LEN * DM);

  // weights: transpose+convert, in [K][N] f32 -> out [N][K] bf16
  auto trT = [&](const float* w, bf16* wt, int K, int N) {
    transpose_f32_bf16<<<dim3(N / 32, K / 32), dim3(32, 8), 0, stream>>>(w, wt, K, N);
  };
  trT(Wq,   WqT,   DM,  DM);
  trT(Wdkv, WdkvT, DM,  LAT);
  trT(Wuk,  WukT,  LAT, DM);
  trT(Wuv,  WuvT,  LAT, DM);
  trT(Wo,   WoT,   DM,  DM);
  for (int w = 0; w < 4; ++w)
    trT(wheelW + (size_t)w * DM * 256, whWT + (size_t)w * 256 * DM, DM, 256);

  // q = emb @ Wq                           [2048,1024]
  gemm_bf16_128x128<<<dim3(DM / 128, S_LEN / 128), 256, 0, stream>>>(
      emb_bf, WqT, S_LEN, DM, DM, nullptr, 0, q_bf, DM, nullptr);
  // c_kv = emb @ Wdkv                      [2048,256]
  gemm_bf16_128x128<<<dim3(LAT / 128, S_LEN / 128), 256, 0, stream>>>(
      emb_bf, WdkvT, S_LEN, LAT, DM, nullptr, 0, ckv_bf, LAT, nullptr);
  // k = c_kv @ Wuk, v = c_kv @ Wuv         [2048,1024]
  gemm_bf16_128x128<<<dim3(DM / 128, S_LEN / 128), 256, 0, stream>>>(
      ckv_bf, WukT, S_LEN, DM, LAT, nullptr, 0, k_bf, DM, nullptr);
  gemm_bf16_128x128<<<dim3(DM / 128, S_LEN / 128), 256, 0, stream>>>(
      ckv_bf, WuvT, S_LEN, DM, LAT, nullptr, 0, v_bf, DM, nullptr);

  // transpose V once ([S][1024] -> [1024][S]); amortized over 32 query tiles
  transpose_bf16<<<dim3(DM / 32, S_LEN / 32), dim3(32, 8), 0, stream>>>(
      v_bf, vt_bf, S_LEN, DM);

  // flash attention: grid = (32 query tiles, 16 heads)
  mla_flash_attn<<<dim3(S_LEN / 64, NH), 128, 0, stream>>>(q_bf, k_bf, vt_bf, ctx_bf);

  // body = ctx @ Wo -> out[:, 0:1024] (f32) + bf16 copy for wheels
  gemm_bf16_128x128<<<dim3(DM / 128, S_LEN / 128), 256, 0, stream>>>(
      ctx_bf, WoT, S_LEN, DM, DM, out, 2048, body_bf, DM, nullptr);

  // wheels: out[:, 1024 + w*256 : ...] = body @ wheel_W[w] + wheel_b[w]
  for (int w = 0; w < 4; ++w) {
    gemm_bf16_128x128<<<dim3(256 / 128, S_LEN / 128), 256, 0, stream>>>(
        body_bf, whWT + (size_t)w * 256 * DM, S_LEN, 256, DM,
        out + 1024 + w * 256, 2048, nullptr, 0, wheelB + w * 256);
  }
}
